// GPT_72009421684907
// MI455X (gfx1250) — matrix operand order repaired and checked by reading, not compiled, not run on hardware
//
#include <hip/hip_runtime.h>
#include <hip/hip_bf16.h>

// GPT-2 small forward (L=4, H=12, C=768, T=1024, V=50257, B=2) for gfx1250.
// Memory-bound: stream fp32 weights once (convert to bf16 in-register),
// bf16 activations, WMMA f32_16x16x32_bf16 everywhere, flash attention.
// GEMM: LDS double-buffered, one barrier per K-step, compile-time specialized
// staging paths (vector for N%128==0 layer GEMMs, scalar for the head).

typedef __bf16 bf16_t;
typedef __attribute__((ext_vector_type(16))) __bf16 v16bf;
typedef __attribute__((ext_vector_type(8)))  __bf16 v8bf;
typedef __attribute__((ext_vector_type(4)))  __bf16 v4bf;
typedef __attribute__((ext_vector_type(8)))  float  v8f;
typedef __attribute__((ext_vector_type(4)))  float  v4f;

union Frag { v16bf v; v8bf h[2]; };

// ---------------------------------------------------------------------------
// WMMA helpers (CDNA5 16x16x32 bf16, f32 accumulate)
// ---------------------------------------------------------------------------
__device__ __forceinline__ v8f wmma_bf16(v16bf a, v16bf b, v8f c) {
  return __builtin_amdgcn_wmma_f32_16x16x32_bf16(
      /*neg_a=*/false, a, /*neg_b=*/false, b,
      /*c_mod=*/(short)0, c, /*reuse_a=*/false, /*reuse_b=*/false);
}

// A-matrix 16x32 bf16 fragment: lane l -> row (l&15); VGPR0-3 hold K = kb..kb+7,
// VGPR4-7 hold K = 16+kb..23+kb, where kb = (l>>4)*8. Rows contiguous (stride ld).
__device__ __forceinline__ v16bf load_fragA(const bf16_t* p, int ld) {
  int lane = threadIdx.x & 31;
  const bf16_t* rp = p + (size_t)(lane & 15) * ld + ((lane >> 4) << 3);
  Frag f;
  f.h[0] = *(const v8bf*)(rp);
  f.h[1] = *(const v8bf*)(rp + 16);
  return f.v;
}

// B-matrix 32x16 bf16 fragment from [N][K] storage: lane l -> column (l&15);
// VGPR0-3 hold K = kb..kb+7, VGPR4-7 hold K = 16+kb..23+kb, kb = (l>>4)*8.
__device__ __forceinline__ v16bf load_fragB(const bf16_t* p, int ld) {
  int lane = threadIdx.x & 31;
  const bf16_t* rp = p + (size_t)(lane & 15) * ld + ((lane >> 4) << 3);
  Frag f;
  f.h[0] = *(const v8bf*)(rp);
  f.h[1] = *(const v8bf*)(rp + 16);
  return f.v;
}

__device__ __forceinline__ float gelu_f(float x) {
  const float k = 0.7978845608028654f;
  return 0.5f * x * (1.f + tanhf(k * (x + 0.044715f * x * x * x)));
}

// ---------------------------------------------------------------------------
// Embedding: x[b,t,:] = wte[idx[b,t],:] + wpe[t,:]
// ---------------------------------------------------------------------------
__global__ void embed_kernel(const int* __restrict__ idx,
                             const float* __restrict__ wte,
                             const float* __restrict__ wpe,
                             float* __restrict__ x) {
  int token = blockIdx.x;          // 0..2047
  int t = token & 1023;
  int id = idx[token];
  const float* we = wte + (size_t)id * 768;
  const float* wp = wpe + (size_t)t * 768;
  float* xr = x + (size_t)token * 768;
  for (int c = threadIdx.x; c < 768; c += blockDim.x)
    xr[c] = we[c] + wp[c];
}

// ---------------------------------------------------------------------------
// LayerNorm: one wave per row of 768; bf16 output for WMMA consumption.
// ---------------------------------------------------------------------------
__global__ __launch_bounds__(256)
void ln_kernel(const float* __restrict__ x,
               const float* __restrict__ w,
               const float* __restrict__ b,
               bf16_t* __restrict__ out) {
  int row = blockIdx.x * 8 + (threadIdx.x >> 5);
  int lane = threadIdx.x & 31;
  const float* xr = x + (size_t)row * 768;
  float vals[24];
  float s = 0.f, ss = 0.f;
#pragma unroll
  for (int i = 0; i < 24; ++i) {
    float v = xr[lane + i * 32];
    vals[i] = v;
    s += v;
    ss += v * v;
  }
#pragma unroll
  for (int off = 16; off > 0; off >>= 1) {
    s  += __shfl_xor(s,  off, 32);
    ss += __shfl_xor(ss, off, 32);
  }
  float mu  = s * (1.f / 768.f);
  float var = ss * (1.f / 768.f) - mu * mu;
  float inv = rsqrtf(var + 1e-5f);
  bf16_t* orow = out + (size_t)row * 768;
#pragma unroll
  for (int i = 0; i < 24; ++i) {
    int c = lane + i * 32;
    orow[c] = (bf16_t)((vals[i] - mu) * inv * w[c] + b[c]);
  }
}

// ---------------------------------------------------------------------------
// GEMM: out[M,N] = A[M,K](bf16) @ W[K,N](f32, converted to bf16 while staging)
// Block tile 128x128x32; 8 waves, each 64x32 (4x2 WMMA tiles).
// Double-buffered LDS: next tile's global loads are issued before the current
// tile's WMMAs, then stored to the alternate buffer; ONE barrier per K-step.
// MODE 0: N multiple of 128 & 4-aligned -> b128 W loads, b64 transposed stores.
// MODE 1: head (odd N) -> scalar W loads, bounds-checked only in the tail strip.
// Epilogue: + bias, optional GELU, optional residual add; fp32 and/or bf16 out.
// ---------------------------------------------------------------------------
#define GLS 40  // LDS row stride (bf16 elems) -> 80B rows, 16B-aligned, pad banks

template <int MODE>
__global__ __launch_bounds__(256)
void gemm_kernel(int M, int N, int K,
                 const bf16_t* __restrict__ A, int lda,
                 const float* __restrict__ W,
                 const float* __restrict__ bias,
                 const float* __restrict__ residual,
                 float* __restrict__ outf, int ldo,
                 bf16_t* __restrict__ outb, int ldob,
                 int do_gelu) {
  __shared__ bf16_t As[2][128 * GLS];
  __shared__ bf16_t Bs[2][128 * GLS];  // transposed: [n][k]

  int n0 = blockIdx.x * 128;
  int m0 = blockIdx.y * 128;
  int tid = threadIdx.x;
  int lane = tid & 31;
  int wv = tid >> 5;
  int waveM = wv >> 2;  // 0..1 -> 64 rows
  int waveN = wv & 3;   // 0..3 -> 32 cols

  // Per-thread staging coordinates.
  const int arow = tid >> 1;            // A: 128 rows x 32 k, 16 elems/thread
  const int acol = (tid & 1) * 16;
  const int kq = (tid >> 5) * 4;        // W vec path: 4k x 4n sub-block
  const int nq = (tid & 31) * 4;
  const int skk = tid >> 3;             // W scalar path: 1k x 16n strip
  const int snn = (tid & 7) * 16;

  v8f acc[4][2];
#pragma unroll
  for (int mt = 0; mt < 4; ++mt)
#pragma unroll
    for (int nt = 0; nt < 2; ++nt)
#pragma unroll
      for (int r = 0; r < 8; ++r) acc[mt][nt][r] = 0.f;

  // Staging registers for the in-flight tile.
  v8bf ra0, ra1;
  v4f  rw0, rw1, rw2, rw3;   // MODE 0
  float rf[16];              // MODE 1

  // ---- load tile 0 into registers ----
  {
    const bf16_t* ap = A + (size_t)(m0 + arow) * lda + acol;
    ra0 = *(const v8bf*)(ap);
    ra1 = *(const v8bf*)(ap + 8);
    if (MODE == 0) {
      const float* wp = W + (size_t)kq * N + n0 + nq;
      rw0 = *(const v4f*)(wp);
      rw1 = *(const v4f*)(wp + N);
      rw2 = *(const v4f*)(wp + 2 * (size_t)N);
      rw3 = *(const v4f*)(wp + 3 * (size_t)N);
    } else {
      const float* wp = W + (size_t)skk * N + n0 + snn;
      if (n0 + snn + 16 <= N) {
#pragma unroll
        for (int i = 0; i < 16; ++i) rf[i] = wp[i];
      } else {
#pragma unroll
        for (int i = 0; i < 16; ++i)
          rf[i] = (n0 + snn + i < N) ? wp[i] : 0.f;
      }
    }
  }
  // ---- store tile 0 to buffer 0 ----
  {
    *(v8bf*)&As[0][arow * GLS + acol]     = ra0;
    *(v8bf*)&As[0][arow * GLS + acol + 8] = ra1;
    if (MODE == 0) {
#pragma unroll
      for (int i = 0; i < 4; ++i) {
        v4bf q = {(bf16_t)rw0[i], (bf16_t)rw1[i], (bf16_t)rw2[i], (bf16_t)rw3[i]};
        *(v4bf*)&Bs[0][(nq + i) * GLS + kq] = q;
      }
    } else {
#pragma unroll
      for (int i = 0; i < 16; ++i)
        Bs[0][(snn + i) * GLS + skk] = (bf16_t)rf[i];
    }
  }
  __syncthreads();

  int cur = 0;
  for (int k0 = 0; k0 < K; k0 += 32) {
    const bool hasNext = (k0 + 32 < K);

    // ---- issue next tile's global loads (overlap with WMMA below) ----
    if (hasNext) {
      const int kn = k0 + 32;
      const bf16_t* ap = A + (size_t)(m0 + arow) * lda + kn + acol;
      ra0 = *(const v8bf*)(ap);
      ra1 = *(const v8bf*)(ap + 8);
      if (MODE == 0) {
        const float* wp = W + (size_t)(kn + kq) * N + n0 + nq;
        rw0 = *(const v4f*)(wp);
        rw1 = *(const v4f*)(wp + N);
        rw2 = *(const v4f*)(wp + 2 * (size_t)N);
        rw3 = *(const v4f*)(wp + 3 * (size_t)N);
      } else {
        const float* wp = W + (size_t)(kn + skk) * N + n0 + snn;
        if (n0 + snn + 16 <= N) {
#pragma unroll
          for (int i = 0; i < 16; ++i) rf[i] = wp[i];
        } else {
#pragma unroll
          for (int i = 0; i < 16; ++i)
            rf[i] = (n0 + snn + i < N) ? wp[i] : 0.f;
        }
      }
      if (k0 + 64 < K)  // prefetch one tile further ahead
        __builtin_prefetch(W + (size_t)(k0 + 64 + skk) * N + n0 + snn, 0, 1);
    }

    // ---- compute from buffer `cur` ----
    v16bf af[4], bfrag[2];
#pragma unroll
    for (int mt = 0; mt < 4; ++mt)
      af[mt] = load_fragA(&As[cur][(waveM * 64 + mt * 16) * GLS], GLS);
#pragma unroll
    for (int nt = 0; nt < 2; ++nt)
      bfrag[nt] = load_fragB(&Bs[cur][(waveN * 32 + nt * 16) * GLS], GLS);
#pragma unroll
    for (int mt = 0; mt < 4; ++mt)
#pragma unroll
      for (int nt = 0; nt < 2; ++nt)
        acc[mt][nt] = wmma_bf16(af[mt], bfrag[nt], acc[mt][nt]);

    // ---- store next tile to the alternate buffer, single barrier ----
    if (hasNext) {
      int nxt = cur ^ 1;
      *(v8bf*)&As[nxt][arow * GLS + acol]     = ra0;
      *(v8bf*)&As[nxt][arow * GLS + acol + 8] = ra1;
      if (MODE == 0) {
#pragma unroll
        for (int i = 0; i < 4; ++i) {
          v4bf q = {(bf16_t)rw0[i], (bf16_t)rw1[i], (bf16_t)rw2[i], (bf16_t)rw3[i]};
          *(v4bf*)&Bs[nxt][(nq + i) * GLS + kq] = q;
        }
      } else {
#pragma unroll
        for (int i = 0; i < 16; ++i)
          Bs[nxt][(snn + i) * GLS + skk] = (bf16_t)rf[i];
      }
      __syncthreads();
      cur = nxt;
    }
  }

  // Epilogue. C/D layout: lane -> col (lane&15), VGPR r -> row r + 8*(lane>>4).
  int l15 = lane & 15;
  int half = lane >> 4;
#pragma unroll
  for (int mt = 0; mt < 4; ++mt) {
#pragma unroll
    for (int nt = 0; nt < 2; ++nt) {
      int col = n0 + waveN * 32 + nt * 16 + l15;
      if (col >= N) continue;
      float bb = bias ? bias[col] : 0.f;
      int rbase = m0 + waveM * 64 + mt * 16 + half * 8;
#pragma unroll
      for (int r = 0; r < 8; ++r) {
        int row = rbase + r;
        float v = acc[mt][nt][r] + bb;
        if (do_gelu) v = gelu_f(v);
        if (residual) v += residual[(size_t)row * ldo + col];
        if (outf) outf[(size_t)row * ldo + col] = v;
        if (outb) outb[(size_t)row * ldob + col] = (bf16_t)v;
      }
    }
  }
}

// ---------------------------------------------------------------------------
// Flash attention: workgroup = (qblock of 128 rows, head, batch); 8 waves,
// each wave owns 16 query rows. qkv layout [B,T,3C] bf16; head dim D=64.
// K and V blocks staged cooperatively in LDS (K row-major: B-fragments read
// contiguous dims per key; V transposed for P@V). Online softmax with fully
// unrolled 16-lane shuffles; mask applied only on diagonal blocks.
// ---------------------------------------------------------------------------
#define SV 40
#define SP 40
#define SK 72   // K LDS row stride (bf16): 144B rows, 16B aligned, bank-padded

// Online-softmax update for one 16x32 S tile pair held in (s0, s1).
template <bool MASKED>
__device__ __forceinline__ void softmax_update(
    v8f& s0, v8f& s1, v8f o[4], float* mv, float* lv, bf16_t* pl,
    int k0, int qrow0, int l15, int half) {
#pragma unroll
  for (int r = 0; r < 8; ++r) {
    float v0 = s0[r] * 0.125f;  // 1/sqrt(64)
    float v1 = s1[r] * 0.125f;
    if (MASKED) {
      int row = qrow0 + r + half * 8;
      if (k0 + l15 > row)      v0 = -1e30f;
      if (k0 + 16 + l15 > row) v1 = -1e30f;
    }
    float rm = fmaxf(v0, v1);
#pragma unroll
    for (int off = 1; off < 16; off <<= 1)
      rm = fmaxf(rm, __shfl_xor(rm, off, 32));
    float mn = fmaxf(mv[r], rm);
    float c  = __expf(mv[r] - mn);
    float p0 = __expf(v0 - mn);
    float p1 = __expf(v1 - mn);
    float rs = p0 + p1;
#pragma unroll
    for (int off = 1; off < 16; off <<= 1)
      rs += __shfl_xor(rs, off, 32);
    lv[r] = lv[r] * c + rs;
    mv[r] = mn;
    o[0][r] *= c; o[1][r] *= c; o[2][r] *= c; o[3][r] *= c;
    int rr = r + half * 8;
    pl[rr * SP + l15]      = (bf16_t)p0;
    pl[rr * SP + 16 + l15] = (bf16_t)p1;
  }
}

__global__ __launch_bounds__(256)
void attn_kernel(const bf16_t* __restrict__ qkv, bf16_t* __restrict__ y) {
  __shared__ bf16_t Ks[32 * SK];        // [key][dim] current 32-key block
  __shared__ bf16_t Vt[64 * SV];        // [dim][key] current 32-key block
  __shared__ bf16_t Pl[8 * 16 * SP];    // per-wave P tile [16 q][32 k]

  int qblk = blockIdx.x;  // 0..7
  int h    = blockIdx.y;  // 0..11
  int b    = blockIdx.z;  // 0..1
  int tid  = threadIdx.x;
  int lane = tid & 31;
  int wv   = tid >> 5;
  int l15  = lane & 15;
  int half = lane >> 4;
  int qrow0 = qblk * 128 + wv * 16;

  const bf16_t* qb = qkv + ((size_t)(b * 1024 + qrow0)) * 2304 + h * 64;
  v16bf aq0 = load_fragA(qb, 2304);        // head dims 0..31
  v16bf aq1 = load_fragA(qb + 32, 2304);   // head dims 32..63

  v8f o[4];
#pragma unroll
  for (int dt = 0; dt < 4; ++dt)
#pragma unroll
    for (int r = 0; r < 8; ++r) o[dt][r] = 0.f;
  float mv[8], lv[8];
#pragma unroll
  for (int r = 0; r < 8; ++r) { mv[r] = -1e30f; lv[r] = 0.f; }

  // K-staging: all 256 threads copy one 8-dim chunk (32 keys x 64 dims).
  const int kkey = tid >> 3;
  const int kseg = (tid & 7) * 8;
  // V-staging: threads 0..127 each own a 4key x 4dim sub-block (transpose).
  const int vkey = (tid & 7) * 4;
  const int vdim = ((tid >> 3) & 15) * 4;
  const bool vstager = (tid < 128);

  int nkb = (qblk + 1) * 4;  // causal: key blocks of 32 up to qrow_max
  for (int kb = 0; kb < nkb; ++kb) {
    int k0 = kb * 32;

    // Cooperative K stage: K[32 keys][64 dims] -> Ks[key][dim] (row-major).
    {
      const bf16_t* kr =
          qkv + ((size_t)(b * 1024 + k0 + kkey)) * 2304 + 768 + h * 64 + kseg;
      *(v8bf*)&Ks[kkey * SK + kseg] = *(const v8bf*)kr;
    }
    // Cooperative V stage: V[32 keys][64 dims] -> Vt[dim][key], vectorized.
    if (vstager) {
      const bf16_t* vr =
          qkv + ((size_t)(b * 1024 + k0 + vkey)) * 2304 + 1536 + h * 64 + vdim;
      v4bf r0 = *(const v4bf*)(vr);
      v4bf r1 = *(const v4bf*)(vr + 2304);
      v4bf r2 = *(const v4bf*)(vr + 2 * 2304);
      v4bf r3 = *(const v4bf*)(vr + 3 * 2304);
#pragma unroll
      for (int i = 0; i < 4; ++i) {
        v4bf q = {r0[i], r1[i], r2[i], r3[i]};
        *(v4bf*)&Vt[(vdim + i) * SV + vkey] = q;
      }
    }
    __syncthreads();

    bool active = (k0 <= qrow0 + 15);
    if (active) {
      v8f s0, s1;
#pragma unroll
      for (int r = 0; r < 8; ++r) { s0[r] = 0.f; s1[r] = 0.f; }
      v16bf b00 = load_fragB(&Ks[0], SK);             // keys 0..15, dims 0..31
      v16bf b01 = load_fragB(&Ks[32], SK);            // keys 0..15, dims 32..63
      v16bf b10 = load_fragB(&Ks[16 * SK], SK);       // keys 16..31, dims 0..31
      v16bf b11 = load_fragB(&Ks[16 * SK + 32], SK);  // keys 16..31, dims 32..63
      s0 = wmma_bf16(aq0, b00, s0);
      s0 = wmma_bf16(aq1, b01, s0);
      s1 = wmma_bf16(aq0, b10, s1);
      s1 = wmma_bf16(aq1, b11, s1);

      bf16_t* pl = &Pl[wv * 16 * SP];
      if (k0 + 31 <= qrow0) {
        softmax_update<false>(s0, s1, o, mv, lv, pl, k0, qrow0, l15, half);
      } else {
        softmax_update<true>(s0, s1, o, mv, lv, pl, k0, qrow0, l15, half);
      }

      v16bf ap = load_fragA(pl, SP);  // P as A-matrix [16 q][32 keys]
#pragma unroll
      for (int dt = 0; dt < 4; ++dt) {
        v16bf bv = load_fragB(&Vt[(dt * 16) * SV], SV);
        o[dt] = wmma_bf16(ap, bv, o[dt]);
      }
    }
    __syncthreads();
  }

#pragma unroll
  for (int dt = 0; dt < 4; ++dt) {
    int col = h * 64 + dt * 16 + l15;
#pragma unroll
    for (int r = 0; r < 8; ++r) {
      int row = qrow0 + r + half * 8;
      y[(size_t)(b * 1024 + row) * 768 + col] = (bf16_t)(o[dt][r] / lv[r]);
    }
  }
}

// ---------------------------------------------------------------------------
// Host launcher
// ---------------------------------------------------------------------------
extern "C" void kernel_launch(void* const* d_in, const int* in_sizes, int n_in,
                              void* d_out, int out_size, void* d_ws, size_t ws_size,
                              hipStream_t stream) {
  const int*   idx    = (const int*)  d_in[0];
  const float* wte    = (const float*)d_in[1];
  const float* wpe    = (const float*)d_in[2];
  const float* ln1_w  = (const float*)d_in[3];
  const float* ln1_b  = (const float*)d_in[4];
  const float* qkv_w  = (const float*)d_in[5];
  const float* qkv_b  = (const float*)d_in[6];
  const float* proj_w = (const float*)d_in[7];
  const float* proj_b = (const float*)d_in[8];
  const float* ln2_w  = (const float*)d_in[9];
  const float* ln2_b  = (const float*)d_in[10];
  const float* fc_w   = (const float*)d_in[11];
  const float* fc_b   = (const float*)d_in[12];
  const float* fc2_w  = (const float*)d_in[13];
  const float* fc2_b  = (const float*)d_in[14];
  const float* lnf_w  = (const float*)d_in[15];
  const float* lnf_b  = (const float*)d_in[16];
  const float* head_w = (const float*)d_in[17];
  float* logits = (float*)d_out;

  const int M = 2048, C = 768, V = 50257;

  // Workspace carve-up (~33 MB).
  char* ws = (char*)d_ws;
  size_t off = 0;
  float*  x    = (float*) (ws + off); off += (size_t)M * C * 4;       // residual fp32
  bf16_t* hbuf = (bf16_t*)(ws + off); off += (size_t)M * C * 2;       // LN out bf16
  bf16_t* qkvb = (bf16_t*)(ws + off); off += (size_t)M * 3 * C * 2;   // qkv bf16
  bf16_t* ybuf = (bf16_t*)(ws + off); off += (size_t)M * C * 2;       // attn out bf16
  bf16_t* actb = (bf16_t*)(ws + off); off += (size_t)M * 4 * C * 2;   // gelu(fc) bf16

  embed_kernel<<<M, 256, 0, stream>>>(idx, wte, wpe, x);

  for (int l = 0; l < 4; ++l) {
    ln_kernel<<<M / 8, 256, 0, stream>>>(x, ln1_w + l * C, ln1_b + l * C, hbuf);

    // qkv = ln1(x) @ qkv_w + qkv_b  -> bf16 [M, 3C]
    gemm_kernel<0><<<dim3(3 * C / 128, M / 128), 256, 0, stream>>>(
        M, 3 * C, C, hbuf, C, qkv_w + (size_t)l * C * 3 * C,
        qkv_b + (size_t)l * 3 * C, nullptr, nullptr, 0, qkvb, 3 * C, 0);

    attn_kernel<<<dim3(8, 12, 2), 256, 0, stream>>>(qkvb, ybuf);

    // x += y @ proj_w + proj_b  (residual epilogue, fp32 in-place)
    gemm_kernel<0><<<dim3(C / 128, M / 128), 256, 0, stream>>>(
        M, C, C, ybuf, C, proj_w + (size_t)l * C * C,
        proj_b + (size_t)l * C, x, x, C, nullptr, 0, 0);

    ln_kernel<<<M / 8, 256, 0, stream>>>(x, ln2_w + l * C, ln2_b + l * C, hbuf);

    // act = gelu(ln2(x) @ fc_w + fc_b) -> bf16 [M, 4C]
    gemm_kernel<0><<<dim3(4 * C / 128, M / 128), 256, 0, stream>>>(
        M, 4 * C, C, hbuf, C, fc_w + (size_t)l * C * 4 * C,
        fc_b + (size_t)l * 4 * C, nullptr, nullptr, 0, actb, 4 * C, 1);

    // x += act @ fc2_w + fc2_b
    gemm_kernel<0><<<dim3(C / 128, M / 128), 256, 0, stream>>>(
        M, C, 4 * C, actb, 4 * C, fc2_w + (size_t)l * 4 * C * C,
        fc2_b + (size_t)l * C, x, x, C, nullptr, 0, 0);
  }

  ln_kernel<<<M / 8, 256, 0, stream>>>(x, lnf_w, lnf_b, hbuf);

  // logits = lnf(x) @ head_w  -> fp32 [M, V], V odd -> scalar-staging variant
  gemm_kernel<1><<<dim3((V + 127) / 128, M / 128), 256, 0, stream>>>(
      M, V, C, hbuf, C, head_w, nullptr, nullptr, logits, V, nullptr, 0, 0);
}
